// MultiLayerRGTN_48773648613824
// MI455X (gfx1250) — compile-verified, matrix operand order checked by reading, not run on hardware
//
#include <hip/hip_runtime.h>
#include <math.h>

#define NN 100000
#define EE 1600000
#define CCH 64
#define HH 8
#define DDH 8
#define RR 4
#define LLAY 2

typedef __bf16 bf16_t;
typedef __attribute__((ext_vector_type(16))) __bf16 v16bf;
typedef __attribute__((ext_vector_type(8)))  float  v8f;
typedef __attribute__((ext_vector_type(4)))  unsigned int uint4v;
typedef __attribute__((ext_vector_type(4)))  float  float4v;
typedef __attribute__((ext_vector_type(2)))  float  float2v;

union Frag { uint4v u[2]; v16bf v; };

__device__ __forceinline__ bf16_t f2bf(float f) {
  unsigned u = __builtin_bit_cast(unsigned, f);
  u += 0x7FFFu + ((u >> 16) & 1u);                 // round-to-nearest-even
  unsigned short s = (unsigned short)(u >> 16);
  return __builtin_bit_cast(bf16_t, s);
}

// ---------------------------------------------------------------------------
// Pre-swizzle 34 weight matrices (64x64 each) into WMMA B-fragment layout:
//   frag[mat][kb(2)][tn(4)][lane(32)][j(16)] , K(j,half)= kb*32 + half*8 + (j<8?j:j+8)
// ---------------------------------------------------------------------------
__global__ void prep_weights(const float* __restrict__ WQ, const float* __restrict__ WK,
                             const float* __restrict__ WV, const float* __restrict__ Wcat,
                             const float* __restrict__ Wself, bf16_t* __restrict__ frags) {
  int mat = blockIdx.x;            // 0..33 ; 17 per layer: [r0 QKV .. r3 QKV][cat r0..3][self]
  int layer = mat / 17;
  int m = mat % 17;
  bf16_t* dst = frags + (size_t)mat * 4096;
  for (int eidx = threadIdx.x; eidx < 4096; eidx += blockDim.x) {
    int j    = eidx & 15;
    int lane = (eidx >> 4) & 31;
    int tn   = (eidx >> 9) & 3;
    int kb   = eidx >> 11;
    int half = lane >> 4;
    int n    = tn * 16 + (lane & 15);              // output column
    int K    = kb * 32 + half * 8 + (j < 8 ? j : j + 8);  // reduction index
    float w;
    if (m < 12) {                                  // Q/K/V: einsum('nc,hcd->nhd'), col = h*8+d
      int r = m / 3, which = m % 3;
      const float* W = which == 0 ? WQ : (which == 1 ? WK : WV);
      int h = n >> 3, d = n & 7;
      w = W[((((size_t)layer * RR + r) * HH + h) * CCH + K) * DDH + d];
    } else if (m < 16) {                           // W_concat[l,r,HD,C]
      int r = m - 12;
      w = Wcat[(((size_t)layer * RR + r) * (HH * DDH) + K) * CCH + n];
    } else {                                       // W_self[l,C,C]
      w = Wself[((size_t)layer * CCH + K) * CCH + n];
    }
    dst[eidx] = f2bf(w);
  }
}

__global__ void to_bf16_kernel(const float* __restrict__ x, bf16_t* __restrict__ xb, int total) {
  int i = blockIdx.x * blockDim.x + threadIdx.x;
  if (i < total) xb[i] = f2bf(x[i]);
}

// ---------------------------------------------------------------------------
// GEMM: Out(N x 16*n_ct) = A(N x 64, bf16) @ Wfrags  — one wave per 16x16 tile,
// K=64 via two v_wmma_f32_16x16x32_bf16. blockDim=128 (4 waves).
// ---------------------------------------------------------------------------
__global__ void gemm_bf16_wmma(const bf16_t* __restrict__ A,
                               const bf16_t* __restrict__ frags,
                               float* __restrict__ Out, int ldOut, int n_ct) {
  int wave = threadIdx.x >> 5;
  int lane = threadIdx.x & 31;
  int ct = blockIdx.y * 4 + wave;
  if (ct >= n_ct) return;                          // uniform per wave: EXEC stays all-1
  int half = lane >> 4;
  int row  = blockIdx.x * 16 + (lane & 15);
  const bf16_t* ar = A + (size_t)row * 64;
  Frag a0, a1, b0, b1;
  a0.u[0] = *(const uint4v*)(ar + half * 8);        // K  0.. 7 | 8..15
  a0.u[1] = *(const uint4v*)(ar + 16 + half * 8);   // K 16..23 | 24..31
  a1.u[0] = *(const uint4v*)(ar + 32 + half * 8);
  a1.u[1] = *(const uint4v*)(ar + 48 + half * 8);
  int matl = ct >> 2, tn = ct & 3;
  const bf16_t* bb = frags + (size_t)matl * 4096;
  const bf16_t* p0 = bb + ((size_t)tn * 32 + lane) * 16;
  const bf16_t* p1 = bb + 2048 + ((size_t)tn * 32 + lane) * 16;
  b0.u[0] = *(const uint4v*)p0;       b0.u[1] = *(const uint4v*)(p0 + 8);
  b1.u[0] = *(const uint4v*)p1;       b1.u[1] = *(const uint4v*)(p1 + 8);
  v8f c = {0.f, 0.f, 0.f, 0.f, 0.f, 0.f, 0.f, 0.f};
  c = __builtin_amdgcn_wmma_f32_16x16x32_bf16(false, a0.v, false, b0.v, (short)0, c, false, false);
  c = __builtin_amdgcn_wmma_f32_16x16x32_bf16(false, a1.v, false, b1.v, (short)0, c, false, false);
  int m0  = blockIdx.x * 16 + half * 8;            // C/D layout: VGPR v -> row half*8+v
  int col = ct * 16 + (lane & 15);
#pragma unroll
  for (int v = 0; v < 8; ++v)
    Out[(size_t)(m0 + v) * ldOut + col] = c[v];
}

// ---------------------------------------------------------------------------
// Fused edge pass (one relation): thread = (edge, head).
// energy -> atomicAdd alpha_n[dst,h];  V[src,h,:] -> atomicAdd S[dst,h,:]
// (alpha factors out of the V segment-sum: m = alpha[dst] * S)
// ---------------------------------------------------------------------------
__global__ void edge_pass(const int* __restrict__ ei, const int* __restrict__ et,
                          const float* __restrict__ QKV,
                          float* __restrict__ alpha, float* __restrict__ S, int r) {
  int idx = blockIdx.x * blockDim.x + threadIdx.x;
  if (idx >= EE * 8) return;
  int e = idx >> 3, h = idx & 7;
  if (et[e] != r) return;
  int src = ei[e], dst = ei[EE + e];
  const float4v* q = (const float4v*)(QKV + (size_t)dst * 192 + h * 8);
  const float4v* k = (const float4v*)(QKV + (size_t)src * 192 + 64 + h * 8);
  const float4v* v = (const float4v*)(QKV + (size_t)src * 192 + 128 + h * 8);
  float4v q0 = q[0], q1 = q[1], k0 = k[0], k1 = k[1], v0 = v[0], v1 = v[1];
  float en = q0.x * k0.x + q0.y * k0.y + q0.z * k0.z + q0.w * k0.w
           + q1.x * k1.x + q1.y * k1.y + q1.z * k1.z + q1.w * k1.w;
  en *= 0.35355339059327373f;                      // 1/sqrt(D)
  en = en > 0.f ? en : 0.01f * en;                 // leaky_relu
  atomicAdd(alpha + (size_t)dst * 8 + h, en);
  float* s = S + (size_t)dst * 64 + h * 8;
  atomicAdd(s + 0, v0.x); atomicAdd(s + 1, v0.y); atomicAdd(s + 2, v0.z); atomicAdd(s + 3, v0.w);
  atomicAdd(s + 4, v1.x); atomicAdd(s + 5, v1.y); atomicAdd(s + 6, v1.z); atomicAdd(s + 7, v1.w);
}

// Per-head global (node-axis) softmax stats: one block per head.
__global__ void softmax_stats(const float* __restrict__ alpha, float* __restrict__ stats) {
  int h = blockIdx.x;
  __shared__ float red[256];
  float mx = -1e30f;
  for (int i = threadIdx.x; i < NN; i += 256) mx = fmaxf(mx, alpha[(size_t)i * 8 + h]);
  red[threadIdx.x] = mx; __syncthreads();
  for (int s = 128; s > 0; s >>= 1) {
    if (threadIdx.x < s) red[threadIdx.x] = fmaxf(red[threadIdx.x], red[threadIdx.x + s]);
    __syncthreads();
  }
  mx = red[0]; __syncthreads();
  float sm = 0.f;
  for (int i = threadIdx.x; i < NN; i += 256) sm += expf(alpha[(size_t)i * 8 + h] - mx);
  red[threadIdx.x] = sm; __syncthreads();
  for (int s = 128; s > 0; s >>= 1) {
    if (threadIdx.x < s) red[threadIdx.x] += red[threadIdx.x + s];
    __syncthreads();
  }
  if (threadIdx.x == 0) { stats[h] = mx; stats[8 + h] = red[0]; }
}

// m[n, h*8+d] = softmax_alpha[n,h] * S[n,h,d]  -> bf16 (input of Wcat GEMM)
__global__ void msg_kernel(const float* __restrict__ alpha, const float* __restrict__ stats,
                           const float* __restrict__ S, bf16_t* __restrict__ mb) {
  int idx = blockIdx.x * blockDim.x + threadIdx.x;
  if (idx >= NN * 64) return;
  int n = idx >> 6, c = idx & 63, h = c >> 3;
  float a = expf(alpha[(size_t)n * 8 + h] - stats[h]) / stats[8 + h];
  mb[idx] = f2bf(a * S[idx]);
}

// ---------------------------------------------------------------------------
// Epilogue: semantic softmax over R, GELU residual, LayerNorm.
// One wave (32 lanes) per node; 2 channels/lane; wave32 shfl reductions.
// ---------------------------------------------------------------------------
__global__ void final_kernel(const float* __restrict__ x, const float* __restrict__ stacked,
                             const float* __restrict__ selfb,
                             const float* __restrict__ sem_w, const float* __restrict__ sem_b,
                             const float* __restrict__ ln_g, const float* __restrict__ ln_b,
                             float* __restrict__ out, int layer) {
  int lane = threadIdx.x & 31;
  int node = blockIdx.x * 8 + (threadIdx.x >> 5);
  if (node >= NN) return;
  int c0 = lane * 2;
  float st[RR][2];
#pragma unroll
  for (int r = 0; r < RR; ++r) {
    float2v s2 = *(const float2v*)(stacked + ((size_t)r * NN + node) * 64 + c0);
    st[r][0] = s2.x; st[r][1] = s2.y;
  }
  float2v sw = *(const float2v*)(sem_w + layer * 64 + c0);
  float sb = sem_b[layer];
  float score[RR];
#pragma unroll
  for (int r = 0; r < RR; ++r) {
    float p = st[r][0] * sw.x + st[r][1] * sw.y;
#pragma unroll
    for (int m = 16; m > 0; m >>= 1) p += __shfl_xor(p, m, 32);
    score[r] = p + sb;
  }
  float smax = fmaxf(fmaxf(score[0], score[1]), fmaxf(score[2], score[3]));
  float beta[RR], bsum = 0.f;
#pragma unroll
  for (int r = 0; r < RR; ++r) { beta[r] = expf(score[r] - smax); bsum += beta[r]; }
  float binv = 1.f / bsum;
  float y[2];
#pragma unroll
  for (int t = 0; t < 2; ++t) {
    int c = c0 + t;
    float agg = 0.f;
#pragma unroll
    for (int r = 0; r < RR; ++r) agg += beta[r] * binv * st[r][t];
    float z = selfb[(size_t)node * 64 + c] + agg;
    float g = 0.5f * z * (1.f + erff(z * 0.70710678118654752f));  // exact GELU
    y[t] = x[(size_t)node * 64 + c] + g;
  }
  float s1 = y[0] + y[1], s2 = y[0] * y[0] + y[1] * y[1];
#pragma unroll
  for (int m = 16; m > 0; m >>= 1) { s1 += __shfl_xor(s1, m, 32); s2 += __shfl_xor(s2, m, 32); }
  float mu = s1 * (1.f / 64.f);
  float var = s2 * (1.f / 64.f) - mu * mu;
  float rstd = rsqrtf(var + 1e-5f);
#pragma unroll
  for (int t = 0; t < 2; ++t) {
    int c = c0 + t;
    out[(size_t)node * 64 + c] = (y[t] - mu) * rstd * ln_g[layer * 64 + c] + ln_b[layer * 64 + c];
  }
}

// ---------------------------------------------------------------------------
extern "C" void kernel_launch(void* const* d_in, const int* in_sizes, int n_in,
                              void* d_out, int out_size, void* d_ws, size_t ws_size,
                              hipStream_t stream) {
  (void)in_sizes; (void)n_in; (void)out_size; (void)ws_size;
  const float* x_in  = (const float*)d_in[0];
  const int*   ei    = (const int*)  d_in[1];
  const int*   et    = (const int*)  d_in[2];
  const float* WQ    = (const float*)d_in[3];
  const float* WK    = (const float*)d_in[4];
  const float* WV    = (const float*)d_in[5];
  const float* Wcat  = (const float*)d_in[6];
  const float* semw  = (const float*)d_in[7];
  const float* semb  = (const float*)d_in[8];
  const float* Wself = (const float*)d_in[9];
  const float* lng   = (const float*)d_in[10];
  const float* lnb   = (const float*)d_in[11];
  float* out = (float*)d_out;

  char* ws = (char*)d_ws;
  auto aln = [](size_t v) { return (v + 255) & ~(size_t)255; };
  size_t o = 0;
  bf16_t* frags   = (bf16_t*)(ws + o); o = aln(o + (size_t)34 * 4096 * sizeof(bf16_t));
  bf16_t* xb      = (bf16_t*)(ws + o); o = aln(o + (size_t)NN * 64 * sizeof(bf16_t));
  float*  QKV     = (float*) (ws + o); o = aln(o + (size_t)NN * 192 * sizeof(float));
  float*  alpha   = (float*) (ws + o); o = aln(o + (size_t)NN * 8 * sizeof(float));
  float*  S       = (float*) (ws + o); o = aln(o + (size_t)NN * 64 * sizeof(float));
  bf16_t* mb      = (bf16_t*)(ws + o); o = aln(o + (size_t)NN * 64 * sizeof(bf16_t));
  float*  stacked = (float*) (ws + o); o = aln(o + (size_t)RR * NN * 64 * sizeof(float));
  float*  selfb   = (float*) (ws + o); o = aln(o + (size_t)NN * 64 * sizeof(float));
  float*  stats   = (float*) (ws + o); o = aln(o + 16 * sizeof(float));

  prep_weights<<<34, 256, 0, stream>>>(WQ, WK, WV, Wcat, Wself, frags);

  for (int layer = 0; layer < LLAY; ++layer) {
    const float* xcur = (layer == 0) ? x_in : out;
    to_bf16_kernel<<<(NN * 64 + 255) / 256, 256, 0, stream>>>(xcur, xb, NN * 64);

    for (int r = 0; r < RR; ++r) {
      dim3 gq(NN / 16, 3);   // 12 col-tiles: Q|K|V
      gemm_bf16_wmma<<<gq, 128, 0, stream>>>(xb, frags + ((size_t)layer * 17 + r * 3) * 4096,
                                             QKV, 192, 12);
      hipMemsetAsync(alpha, 0, (size_t)NN * 8 * sizeof(float), stream);
      hipMemsetAsync(S, 0, (size_t)NN * 64 * sizeof(float), stream);
      int tot = EE * 8;
      edge_pass<<<(tot + 255) / 256, 256, 0, stream>>>(ei, et, QKV, alpha, S, r);
      softmax_stats<<<8, 256, 0, stream>>>(alpha, stats);
      msg_kernel<<<(NN * 64 + 255) / 256, 256, 0, stream>>>(alpha, stats, S, mb);
      dim3 gc(NN / 16, 1);
      gemm_bf16_wmma<<<gc, 128, 0, stream>>>(mb, frags + ((size_t)layer * 17 + 12 + r) * 4096,
                                             stacked + (size_t)r * NN * 64, 64, 4);
    }
    dim3 gs(NN / 16, 1);
    gemm_bf16_wmma<<<gs, 128, 0, stream>>>(xb, frags + ((size_t)layer * 17 + 16) * 4096,
                                           selfb, 64, 4);
    final_kernel<<<NN / 8, 256, 0, stream>>>(xcur, stacked, selfb, semw, semb, lng, lnb,
                                             out, layer);
  }
}
